// RowAttentionWithPairBias_15564961481075
// MI455X (gfx1250) — matrix operand order repaired and checked by reading, not compiled, not run on hardware
//
#include <hip/hip_runtime.h>
#include <hip/hip_bf16.h>
#include <stdint.h>

// Shapes (compile-time constants from the reference)
#define BB   2
#define LL   768
#define DD   256
#define DP   128
#define HH   8
#define DHH  32

typedef __attribute__((ext_vector_type(16))) __bf16 v16bf;
typedef __attribute__((ext_vector_type(8)))  __bf16 v8bf;
typedef __attribute__((ext_vector_type(8)))  float  v8f;
typedef __attribute__((ext_vector_type(4)))  float  v4f;

static __device__ __forceinline__ v16bf frag_cat(v8bf lo, v8bf hi) {
  return __builtin_shufflevector(lo, hi, 0,1,2,3,4,5,6,7,8,9,10,11,12,13,14,15);
}
static __device__ __forceinline__ unsigned pack2bf(float a, float b) {
  union { __bf16 h[2]; unsigned u; } x;
  x.h[0] = (__bf16)a; x.h[1] = (__bf16)b;
  return x.u;
}
// Convert 16 fp32 (two v4f pairs per half) to a bf16 A/B fragment
static __device__ __forceinline__ v16bf cvt16(v4f a0, v4f a1, v4f a2, v4f a3) {
  v16bf r;
#pragma unroll
  for (int q = 0; q < 4; ++q) {
    r[q]      = (__bf16)a0[q];
    r[4 + q]  = (__bf16)a1[q];
    r[8 + q]  = (__bf16)a2[q];
    r[12 + q] = (__bf16)a3[q];
  }
  return r;
}

// ---------------------------------------------------------------------------
// 1) Transpose + convert the five 256x256 weight matrices to bf16.
//    wT[mat][n][k] = (bf16) w[mat][k][n]   (so WMMA B-fragments load contiguously)
// ---------------------------------------------------------------------------
__global__ void prep_w_k(const float* __restrict__ wq, const float* __restrict__ wk,
                         const float* __restrict__ wv, const float* __restrict__ wg,
                         const float* __restrict__ wo, __bf16* __restrict__ wT) {
  int mat = blockIdx.x >> 8;
  int n   = blockIdx.x & 255;
  int k   = threadIdx.x;
  const float* w = (mat == 0) ? wq : (mat == 1) ? wk : (mat == 2) ? wv
                   : (mat == 3) ? wg : wo;
  wT[((size_t)mat * 256 + n) * 256 + k] = (__bf16)w[k * 256 + n];
}

// 1b) wpbT[n][k] = (bf16) wpb[k][n] for n<8, zero for n=8..15 (N padding)
__global__ void prep_wpb_k(const float* __restrict__ wpb, __bf16* __restrict__ wpbT) {
  int t = threadIdx.x;
#pragma unroll
  for (int u = 0; u < 8; ++u) {
    int e = t + 256 * u;          // 0 .. 2047
    int n = e >> 7, k = e & 127;
    wpbT[e] = (n < HH) ? (__bf16)wpb[k * HH + n] : (__bf16)0.0f;
  }
}

// ---------------------------------------------------------------------------
// 2) LayerNorm(single) -> bf16, one wave per row.
// ---------------------------------------------------------------------------
__global__ void ln_k(const float* __restrict__ x, const float* __restrict__ g,
                     const float* __restrict__ b, __bf16* __restrict__ xb) {
  int row  = blockIdx.x;
  int lane = threadIdx.x;
  const float* xr = x + (size_t)row * DD;
  float v[8];
  float s = 0.f;
#pragma unroll
  for (int u = 0; u < 8; ++u) { v[u] = xr[lane + 32 * u]; s += v[u]; }
  for (int m = 16; m; m >>= 1) s += __shfl_xor(s, m, 32);
  float mu = s * (1.f / DD);
  float var = 0.f;
#pragma unroll
  for (int u = 0; u < 8; ++u) { float d = v[u] - mu; var += d * d; }
  for (int m = 16; m; m >>= 1) var += __shfl_xor(var, m, 32);
  float rstd = rsqrtf(var * (1.f / DD) + 1e-5f);
#pragma unroll
  for (int u = 0; u < 8; ++u) {
    int c = lane + 32 * u;
    xb[(size_t)row * DD + c] = (__bf16)((v[u] - mu) * rstd * g[c] + b[c]);
  }
}

// ---------------------------------------------------------------------------
// 3) Fused QKVG projections: (B*L,256) x (256,256), bf16 WMMA, fp32 accum.
//    grid=(96,4): x = 16-row tile, y = which matrix (0=q,1=k,2=v,3=g).
//    block=128: 4 waves, each wave owns a 16x64 output tile (4 WMMA accums).
// ---------------------------------------------------------------------------
__global__ void proj_k(const __bf16* __restrict__ xb, const __bf16* __restrict__ wT,
                       const float* __restrict__ bg,
                       __bf16* __restrict__ qb, __bf16* __restrict__ kbuf,
                       __bf16* __restrict__ vT, float* __restrict__ gv) {
  int r0   = blockIdx.x * 16;
  int mat  = blockIdx.y;
  int wave = threadIdx.x >> 5;
  int lane = threadIdx.x & 31;
  int lh = lane & 15, hi = lane >> 4;
  int kb = hi * 8;
  const __bf16* wmat = wT + (size_t)mat * 256 * 256;
  int ngrp = wave * 64;
  v8f acc[4] = {};
  for (int kk = 0; kk < DD; kk += 32) {
    const __bf16* arow = xb + (size_t)(r0 + lh) * DD + kk;
    v16bf afrag = frag_cat(*(const v8bf*)(arow + kb), *(const v8bf*)(arow + 16 + kb));
#pragma unroll
    for (int t = 0; t < 4; ++t) {
      int n = ngrp + t * 16 + lh;
      const __bf16* brow = wmat + (size_t)n * DD + kk + kb;
      v16bf bfrag = frag_cat(*(const v8bf*)brow, *(const v8bf*)(brow + 16));
      acc[t] = __builtin_amdgcn_wmma_f32_16x16x32_bf16(false, afrag, false, bfrag,
                                                       (short)0, acc[t], false, false);
    }
  }
#pragma unroll
  for (int t = 0; t < 4; ++t) {
    int n = ngrp + t * 16 + lh;
    int h = n >> 5, d = n & 31;
#pragma unroll
    for (int r = 0; r < 8; ++r) {
      int grow = r0 + r + hi * 8;
      int b = grow / LL, i = grow % LL;
      float val = acc[t][r];
      if (mat == 0)      qb[((size_t)(b * HH + h) * LL + i) * DHH + d]  = (__bf16)val;
      else if (mat == 1) kbuf[((size_t)(b * HH + h) * LL + i) * DHH + d] = (__bf16)val;
      else if (mat == 2) vT[((size_t)(b * HH + h) * DHH + d) * LL + i]  = (__bf16)val;
      else               gv[(size_t)grow * DD + n] = 1.f / (1.f + __expf(-(val + bg[n])));
    }
  }
}

// ---------------------------------------------------------------------------
// 4) Pair bias on the matrix cores: bias[b,h,i,j] = sum_p pair[b,i,j,p]*wpb[p,h].
//    This kernel reads the dominant 604MB `pair` tensor exactly once; WMMA
//    (M=16 flat (b,i,j) rows, K=128 channels, N=16 padded heads) replaces
//    ~16K scalar FMAs/tile with 4 WMMAs + on-the-fly fp32->bf16 cvt, so the
//    kernel stays on the HBM roofline instead of going VALU-bound.
//    C-fragment: lane = head, rows = consecutive j -> each active lane stores
//    its whole v8f accumulator contiguously into the (B,H,L,L) bias layout.
// ---------------------------------------------------------------------------
__global__ void bias_k(const float* __restrict__ pair, const __bf16* __restrict__ wpbT,
                       float* __restrict__ bias) {
  int wave = threadIdx.x >> 5;
  int lane = threadIdx.x & 31;
  int lh = lane & 15, hi = lane >> 4, kb = hi * 8;
  long tile = (long)blockIdx.x * 8 + wave;     // 16-row tile of flat (b*L+i)*L+j
  long row0 = tile * 16;

  // B-fragments (wpbT is 4KB, L2-resident): lane holds head n=lh, K=p
  v16bf bf[4];
#pragma unroll
  for (int c = 0; c < 4; ++c) {
    const __bf16* brow = wpbT + (size_t)lh * DP + c * 32 + kb;
    bf[c] = frag_cat(*(const v8bf*)brow, *(const v8bf*)(brow + 16));
  }

  const float* pr = pair + (row0 + lh) * DP;   // this lane's pair row (512B)
  v8f acc = {};
#pragma unroll
  for (int c = 0; c < 4; ++c) {
    int kk = c * 32;
    v4f a0 = *(const v4f*)(pr + kk + kb);
    v4f a1 = *(const v4f*)(pr + kk + kb + 4);
    v4f a2 = *(const v4f*)(pr + kk + 16 + kb);
    v4f a3 = *(const v4f*)(pr + kk + 16 + kb + 4);
    v16bf afrag = cvt16(a0, a1, a2, a3);
    acc = __builtin_amdgcn_wmma_f32_16x16x32_bf16(false, afrag, false, bf[c],
                                                  (short)0, acc, false, false);
  }

  // Decompose row0 -> (b, i, j0); rows of the tile are consecutive j.
  int  j0 = (int)(row0 % LL);
  long t2 = row0 / LL;
  int  i  = (int)(t2 % LL);
  int  b  = (int)(t2 / LL);
  int  h  = lh;                                 // C-layout: lane column = head
  if (h < HH) {
    float* dst = bias + ((long)(b * HH + h) * LL + i) * LL + j0 + kb;
    *(v8f*)dst = acc;                           // 8 consecutive j per lane
  }
}

// ---------------------------------------------------------------------------
// 5) Flash attention per (b,h, 16-row i-tile), one wave per tile.
//    Computes S^T = K * Q^T so the C-layout lane index == query row i:
//    softmax state (m, l) is per-lane; O^T = V^T * P^T keeps lane == i too.
// ---------------------------------------------------------------------------
__global__ void attn_k(const __bf16* __restrict__ qb, const __bf16* __restrict__ kbuf,
                       const __bf16* __restrict__ vT, const float* __restrict__ gv,
                       const float* __restrict__ bias, const unsigned char* __restrict__ mask,
                       __bf16* __restrict__ gated) {
  int bh = blockIdx.x;
  int b = bh >> 3, h = bh & 7;
  int wave = threadIdx.x >> 5;
  int lane = threadIdx.x & 31;
  int lh = lane & 15, hi = lane >> 4, kb = hi * 8;
  int itile = blockIdx.y * 64 + wave * 16;
  int i = itile + lh;

  const __bf16* qrow = qb + ((size_t)bh * LL + i) * DHH + kb;
  v16bf qfrag = frag_cat(*(const v8bf*)qrow, *(const v8bf*)(qrow + 16));

  v8f acc0 = {}, acc1 = {};
  float mrun = -INFINITY, srun = 0.f;
  const float scale = 0.17677669529663687f;  // 1/sqrt(32)
  const float* brow = bias + ((long)bh * LL + i) * LL;
  const unsigned char* mrow = mask + b * LL;

  for (int jb = 0; jb < LL; jb += 32) {
    const __bf16* k0 = kbuf + ((size_t)bh * LL + jb + lh) * DHH;
    const __bf16* k1 = kbuf + ((size_t)bh * LL + jb + 16 + lh) * DHH;
    v16bf ka0 = frag_cat(*(const v8bf*)(k0 + kb), *(const v8bf*)(k0 + 16 + kb));
    v16bf ka1 = frag_cat(*(const v8bf*)(k1 + kb), *(const v8bf*)(k1 + 16 + kb));
    v8f z = {};
    v8f s0 = __builtin_amdgcn_wmma_f32_16x16x32_bf16(false, ka0, false, qfrag, (short)0, z, false, false);
    v8f s1 = __builtin_amdgcn_wmma_f32_16x16x32_bf16(false, ka1, false, qfrag, (short)0, z, false, false);

    v8f b0 = *(const v8f*)(brow + jb + kb);
    v8f b1 = *(const v8f*)(brow + jb + 16 + kb);
    uint2 m0 = *(const uint2*)(mrow + jb + kb);
    uint2 m1 = *(const uint2*)(mrow + jb + 16 + kb);
    float sc0[8], sc1[8];
#pragma unroll
    for (int r = 0; r < 8; ++r) {
      unsigned mb0 = ((r < 4 ? (m0.x >> (8 * r)) : (m0.y >> (8 * (r - 4)))) & 0xffu);
      unsigned mb1 = ((r < 4 ? (m1.x >> (8 * r)) : (m1.y >> (8 * (r - 4)))) & 0xffu);
      sc0[r] = mb0 ? (s0[r] * scale + b0[r]) : -1e9f;
      sc1[r] = mb1 ? (s1[r] * scale + b1[r]) : -1e9f;
    }
    float tmax = -INFINITY;
#pragma unroll
    for (int r = 0; r < 8; ++r) tmax = fmaxf(tmax, fmaxf(sc0[r], sc1[r]));
    tmax = fmaxf(tmax, __shfl_xor(tmax, 16, 32));
    float mnew = fmaxf(mrun, tmax);
    float alpha = __expf(mrun - mnew);
    float p0[8], p1[8], psum = 0.f;
#pragma unroll
    for (int r = 0; r < 8; ++r) {
      p0[r] = __expf(sc0[r] - mnew);
      p1[r] = __expf(sc1[r] - mnew);
      psum += p0[r] + p1[r];
    }
    psum += __shfl_xor(psum, 16, 32);
    srun = srun * alpha + psum;
    mrun = mnew;
    acc0 *= alpha;
    acc1 *= alpha;

    unsigned po0[4], po1[4], qo0[4], qo1[4];
#pragma unroll
    for (int t = 0; t < 4; ++t) {
      po0[t] = pack2bf(p0[2 * t], p0[2 * t + 1]);
      po1[t] = pack2bf(p1[2 * t], p1[2 * t + 1]);
    }
#pragma unroll
    for (int t = 0; t < 4; ++t) {
      qo0[t] = (unsigned)__shfl_xor((int)po0[t], 16, 32);
      qo1[t] = (unsigned)__shfl_xor((int)po1[t], 16, 32);
    }
    union { unsigned u[8]; v16bf v; } pf;
#pragma unroll
    for (int t = 0; t < 4; ++t) {
      pf.u[t]     = hi ? qo1[t] : po0[t];
      pf.u[4 + t] = hi ? po1[t] : qo0[t];
    }

    const __bf16* va0p = vT + ((size_t)bh * DHH + lh) * LL + jb;
    const __bf16* va1p = vT + ((size_t)bh * DHH + 16 + lh) * LL + jb;
    v16bf va0 = frag_cat(*(const v8bf*)(va0p + hi * 16), *(const v8bf*)(va0p + hi * 16 + 8));
    v16bf va1 = frag_cat(*(const v8bf*)(va1p + hi * 16), *(const v8bf*)(va1p + hi * 16 + 8));
    acc0 = __builtin_amdgcn_wmma_f32_16x16x32_bf16(false, va0, false, pf.v, (short)0, acc0, false, false);
    acc1 = __builtin_amdgcn_wmma_f32_16x16x32_bf16(false, va1, false, pf.v, (short)0, acc1, false, false);
  }

  float inv = 1.f / srun;
  const float* grow = gv + ((size_t)(b * LL) + i) * DD + h * DHH;
#pragma unroll
  for (int t = 0; t < 2; ++t) {
    v8f o = t ? acc1 : acc0;
    int dbase = t * 16 + kb;
    v8f gg = *(const v8f*)(grow + dbase);
    v8bf st;
#pragma unroll
    for (int r = 0; r < 8; ++r) st[r] = (__bf16)(o[r] * inv * gg[r]);
    *(v8bf*)(gated + ((size_t)(b * LL) + i) * DD + h * DHH + dbase) = st;
  }
}

// ---------------------------------------------------------------------------
// 6) Output projection + residual: out = single + gated @ wo + bo  (WMMA)
// ---------------------------------------------------------------------------
__global__ void out_k(const __bf16* __restrict__ gated, const __bf16* __restrict__ woT,
                      const float* __restrict__ single, const float* __restrict__ bo,
                      float* __restrict__ out) {
  int r0   = blockIdx.x * 16;
  int wave = threadIdx.x >> 5;
  int lane = threadIdx.x & 31;
  int lh = lane & 15, hi = lane >> 4, kb = hi * 8;
  int ngrp = wave * 64;
  v8f acc[4] = {};
  for (int kk = 0; kk < DD; kk += 32) {
    const __bf16* arow = gated + (size_t)(r0 + lh) * DD + kk;
    v16bf afrag = frag_cat(*(const v8bf*)(arow + kb), *(const v8bf*)(arow + 16 + kb));
#pragma unroll
    for (int t = 0; t < 4; ++t) {
      int n = ngrp + t * 16 + lh;
      const __bf16* brow = woT + (size_t)n * DD + kk + kb;
      v16bf bfrag = frag_cat(*(const v8bf*)brow, *(const v8bf*)(brow + 16));
      acc[t] = __builtin_amdgcn_wmma_f32_16x16x32_bf16(false, afrag, false, bfrag,
                                                       (short)0, acc[t], false, false);
    }
  }
#pragma unroll
  for (int t = 0; t < 4; ++t) {
    int n = ngrp + t * 16 + lh;
#pragma unroll
    for (int r = 0; r < 8; ++r) {
      int row = r0 + r + hi * 8;
      out[(size_t)row * DD + n] = single[(size_t)row * DD + n] + acc[t][r] + bo[n];
    }
  }
}

// ---------------------------------------------------------------------------
extern "C" void kernel_launch(void* const* d_in, const int* in_sizes, int n_in,
                              void* d_out, int out_size, void* d_ws, size_t ws_size,
                              hipStream_t stream) {
  const float* single = (const float*)d_in[0];
  const float* pair   = (const float*)d_in[1];
  const unsigned char* mask = (const unsigned char*)d_in[2];
  const float* ln_g = (const float*)d_in[3];
  const float* ln_b = (const float*)d_in[4];
  const float* wq   = (const float*)d_in[5];
  const float* wk   = (const float*)d_in[6];
  const float* wv   = (const float*)d_in[7];
  const float* wg   = (const float*)d_in[8];
  const float* bg   = (const float*)d_in[9];
  const float* wo   = (const float*)d_in[10];
  const float* bo   = (const float*)d_in[11];
  const float* wpb  = (const float*)d_in[12];
  float* out = (float*)d_out;

  char* ws = (char*)d_ws;
  const size_t szXB  = (size_t)BB * LL * DD * 2;           // LN'd x, bf16
  const size_t szWT  = (size_t)5 * 256 * 256 * 2;          // 5 transposed bf16 weights
  const size_t szQ   = (size_t)BB * HH * LL * DHH * 2;     // q bf16 (B,H,L,Dh)
  const size_t szK   = szQ;                                // k bf16
  const size_t szV   = szQ;                                // v^T bf16 (B,H,Dh,L)
  const size_t szG   = (size_t)BB * LL * DD * 4;           // gate fp32
  const size_t szBI  = (size_t)BB * HH * LL * LL * 4;      // bias fp32 (B,H,L,L)
  const size_t szGA  = (size_t)BB * LL * DD * 2;           // gated bf16
  __bf16* xb    = (__bf16*)(ws);
  __bf16* wT    = (__bf16*)(ws + szXB);
  __bf16* qb    = (__bf16*)(ws + szXB + szWT);
  __bf16* kbuf  = (__bf16*)(ws + szXB + szWT + szQ);
  __bf16* vT    = (__bf16*)(ws + szXB + szWT + szQ + szK);
  float*  gv    = (float*) (ws + szXB + szWT + szQ + szK + szV);
  float*  bias  = (float*) (ws + szXB + szWT + szQ + szK + szV + szG);
  __bf16* gated = (__bf16*)(ws + szXB + szWT + szQ + szK + szV + szG + szBI);
  __bf16* wpbT  = (__bf16*)(ws + szXB + szWT + szQ + szK + szV + szG + szBI + szGA);
  __bf16* woT   = wT + (size_t)4 * 256 * 256;

  prep_w_k<<<5 * 256, 256, 0, stream>>>(wq, wk, wv, wg, wo, wT);
  prep_wpb_k<<<1, 256, 0, stream>>>(wpb, wpbT);
  ln_k<<<BB * LL, 32, 0, stream>>>(single, ln_g, ln_b, xb);
  proj_k<<<dim3(BB * LL / 16, 4), 128, 0, stream>>>(xb, wT, bg, qb, kbuf, vT, gv);
  // B*L*L/16 = 73728 16-row tiles, 8 waves per block
  bias_k<<<(BB * LL * LL / 16) / 8, 256, 0, stream>>>(pair, wpbT, bias);
  attn_k<<<dim3(BB * HH, LL / 64), 128, 0, stream>>>(qb, kbuf, vT, gv, bias, mask, gated);
  out_k<<<BB * LL / 16, 128, 0, stream>>>(gated, woT, single, bo, out);
}
